// LlamaAttentionFused_8804682957409
// MI455X (gfx1250) — matrix operand order repaired and checked by reading, not compiled, not run on hardware
//
#include <hip/hip_runtime.h>

// ---------------------------------------------------------------------------
// Llama attention block, MI455X (gfx1250, wave32, WMMA).
//  - all matmuls on v_wmma_f32_16x16x32_bf16 (f32 accumulate)
//  - weights pre-converted to bf16 and pre-transposed once, so GEMM hot loops
//    stage tiles with GLOBAL_LOAD_ASYNC_TO_LDS_B128 (ASYNCcnt) + LDS double
//    buffering; no conversion VALU in the hot loop.
// ---------------------------------------------------------------------------

typedef __bf16 bf16_t;
typedef __attribute__((ext_vector_type(16))) __bf16 v16bf;
typedef __attribute__((ext_vector_type(8)))  float  v8f;
typedef int v4i __attribute__((vector_size(16)));   // matches builtin proto

#define H_    32
#define KV_   8
#define D_    128
#define HID_  4096
#define S_    2048
#define NQKV  6144   // 4096 (q) + 1024 (k) + 1024 (v)

__device__ __forceinline__ v8f wmma_bf16(v16bf a, v16bf b, v8f c) {
  return __builtin_amdgcn_wmma_f32_16x16x32_bf16(false, a, false, b,
                                                 (short)0, c, false, false);
}

// K-permutation within a 32-group so a lane's 16-element A-fragment is one
// contiguous 32B LDS run.  Chunk-granular: 8-element chunk c -> {0,2,1,3}[c].
__device__ __forceinline__ int perm32(int k) {
  return ((k & 8) << 1) | ((k & 16) >> 1) | (k & 7);
}
__device__ __forceinline__ int chunkperm(int c) {  // c in 0..3
  return ((c & 1) << 1) | (c >> 1);
}

// 16-byte global->LDS copy.  Async (ASYNCcnt-tracked, no VGPR round trip) when
// the gfx1250 builtin is available; otherwise b128 load + b128 LDS store.
__device__ __forceinline__ void cp16(void* l, const void* g) {
#if defined(__HIP_DEVICE_COMPILE__) && \
    __has_builtin(__builtin_amdgcn_global_load_async_to_lds_b128)
  __builtin_amdgcn_global_load_async_to_lds_b128(
      (__attribute__((address_space(1))) v4i*)g,
      (__attribute__((address_space(3))) v4i*)l, 0, 0);
#else
  *(uint4*)l = *(const uint4*)g;
#endif
}
__device__ __forceinline__ void async_wait0() {
#if defined(__HIP_DEVICE_COMPILE__)
#if __has_builtin(__builtin_amdgcn_global_load_async_to_lds_b128)
  asm volatile("s_wait_asynccnt 0x0" ::: "memory");
#endif
#endif
}

// ---------------------------------------------------------------------------
// Pre-pass A: x (f32) -> xb (bf16), same layout [2048][4096].
// ---------------------------------------------------------------------------
__global__ __launch_bounds__(256) void convert_x_kernel(
    const float* __restrict__ x, bf16_t* __restrict__ xb)
{
  int idx = blockIdx.x * 256 + threadIdx.x;   // one per element
  xb[idx] = (bf16_t)x[idx];
}

// ---------------------------------------------------------------------------
// Pre-pass B: weight transpose+convert.  in: f32 [K][N]  ->  out: bf16 [N][K].
// 32x32 LDS tile so both global streams are coalesced.
// ---------------------------------------------------------------------------
__global__ __launch_bounds__(256) void w_transpose_kernel(
    const float* __restrict__ in, bf16_t* __restrict__ out, int N, int K)
{
  __shared__ bf16_t tile[32][33];
  const int n0 = blockIdx.x * 32, k0 = blockIdx.y * 32;
  const int tx = threadIdx.x & 31, ty = threadIdx.x >> 5;   // 32 x 8
  #pragma unroll
  for (int i = 0; i < 4; ++i) {
    int k = ty + i * 8;
    tile[k][tx] = (bf16_t)in[(size_t)(k0 + k) * N + n0 + tx];
  }
  __syncthreads();
  #pragma unroll
  for (int i = 0; i < 4; ++i) {
    int n = ty + i * 8;
    out[(size_t)(n0 + n) * K + k0 + tx] = tile[tx][n];
  }
}

// ---------------------------------------------------------------------------
// GEMM: C[M x N] (f32, row stride ldc) = A[M x 4096](bf16) @ Bt[N x 4096](bf16)
// Block tile 128x128, K-step 32, double-buffered LDS, async staging.
// 8 waves; wave owns 2x4 grid of 16x16 WMMA tiles.
// ---------------------------------------------------------------------------
__global__ __launch_bounds__(256) void gemm_bf16_kernel(
    const bf16_t* __restrict__ A, const bf16_t* __restrict__ Bt,
    float* __restrict__ C, int ldc)
{
  __shared__ __align__(16) bf16_t As[2][128][40];   // natural [m][k]
  __shared__ __align__(16) bf16_t Bs[2][128][40];   // natural [n][k]

  const int tid = threadIdx.x;
  const int m0  = blockIdx.y * 128;
  const int n0  = blockIdx.x * 128;

  const int wave = tid >> 5, lane = tid & 31;
  const int n16  = lane & 15, kh = lane >> 4;
  const int wm   = (wave & 3) * 32;
  const int wn   = (wave >> 2) * 64;

  // A/B tile: 128 rows x 4 chunks (8 bf16 each); 512 chunks, 2 per thread.
  auto stageA = [&](int buf, int k0) {
    #pragma unroll
    for (int i = 0; i < 2; ++i) {
      int idx = i * 256 + tid;
      int r = idx >> 2, c = idx & 3;
      cp16(&As[buf][r][c * 8], &A[(m0 + r) * HID_ + k0 + c * 8]);
    }
  };
  auto stageB = [&](int buf, int k0) {
    #pragma unroll
    for (int i = 0; i < 2; ++i) {
      int idx = i * 256 + tid;
      int n = idx >> 2, c = idx & 3;
      cp16(&Bs[buf][n][c * 8], &Bt[(size_t)(n0 + n) * HID_ + k0 + c * 8]);
    }
  };

  v8f z = {0.f,0.f,0.f,0.f,0.f,0.f,0.f,0.f};
  v8f acc[2][4];
  for (int i = 0; i < 2; ++i)
    for (int j = 0; j < 4; ++j) acc[i][j] = z;

  stageA(0, 0);
  stageB(0, 0);
  async_wait0();
  __syncthreads();

  for (int ks = 0; ks < HID_ / 32; ++ks) {
    const int cur = ks & 1;
    if (ks + 1 < HID_ / 32) {               // prefetch next tile into other buf
      stageA(cur ^ 1, (ks + 1) * 32);
      stageB(cur ^ 1, (ks + 1) * 32);
    }
    v16bf af[2], bfm[4];
    #pragma unroll
    for (int t = 0; t < 2; ++t)
      af[t] = *(const v16bf*)&As[cur][wm + t * 16 + n16][kh * 16];
    #pragma unroll
    for (int t = 0; t < 4; ++t)
      bfm[t] = *(const v16bf*)&Bs[cur][wn + t * 16 + n16][kh * 16];
    #pragma unroll
    for (int i = 0; i < 2; ++i)
      #pragma unroll
      for (int j = 0; j < 4; ++j)
        acc[i][j] = wmma_bf16(af[i], bfm[j], acc[i][j]);
    async_wait0();                          // next tile landed (this wave)
    __syncthreads();                        // all waves done reading cur
  }

  #pragma unroll
  for (int i = 0; i < 2; ++i)
    #pragma unroll
    for (int j = 0; j < 4; ++j)
      #pragma unroll
      for (int v = 0; v < 8; ++v) {
        int row = m0 + wm + i * 16 + v + 8 * kh;
        int col = n0 + wn + j * 16 + n16;
        C[(size_t)row * ldc + col] = acc[i][j][v];
      }
}

// ---------------------------------------------------------------------------
// RoPE (f32) + cache scatter + bf16 q/k/v in attention layouts.
// ---------------------------------------------------------------------------
__global__ __launch_bounds__(256) void rope_scatter_kernel(
    const float* __restrict__ qkv, const float* __restrict__ cosT,
    const float* __restrict__ sinT,
    bf16_t* __restrict__ qb, bf16_t* __restrict__ kb, bf16_t* __restrict__ vb,
    float* __restrict__ ck, float* __restrict__ cv)
{
  const int QP = S_ * H_ * 64, KP = S_ * KV_ * 64, VP = S_ * KV_ * D_;
  int idx = blockIdx.x * 256 + threadIdx.x;
  if (idx < QP) {
    int d = idx & 63, t = idx >> 6, h = t & 31, s = t >> 5;
    float tr = qkv[s * NQKV + h * D_ + d];
    float ti = qkv[s * NQKV + h * D_ + 64 + d];
    float c = cosT[s * 64 + d], sn = sinT[s * 64 + d];
    qb[(h * S_ + s) * D_ + d]      = (bf16_t)(tr * c - ti * sn);
    qb[(h * S_ + s) * D_ + 64 + d] = (bf16_t)(tr * sn + ti * c);
  } else if (idx < QP + KP) {
    int i2 = idx - QP;
    int d = i2 & 63, t = i2 >> 6, kv = t & 7, s = t >> 3;
    float tr = qkv[s * NQKV + 4096 + kv * D_ + d];
    float ti = qkv[s * NQKV + 4096 + kv * D_ + 64 + d];
    float c = cosT[s * 64 + d], sn = sinT[s * 64 + d];
    float r0 = tr * c - ti * sn, r1 = tr * sn + ti * c;
    kb[(kv * S_ + s) * D_ + d]      = (bf16_t)r0;
    kb[(kv * S_ + s) * D_ + 64 + d] = (bf16_t)r1;
    int d1 = d + 64;
    ck[((kv * 16 + (d  >> 3)) * S_ + s) * 8 + (d  & 7)] = r0;
    ck[((kv * 16 + (d1 >> 3)) * S_ + s) * 8 + (d1 & 7)] = r1;
  } else if (idx < QP + KP + VP) {
    int i3 = idx - QP - KP;
    int d = i3 & 127, t = i3 >> 7, kv = t & 7, s = t >> 3;
    float val = qkv[s * NQKV + 5120 + kv * D_ + d];
    vb[(kv * S_ + s) * D_ + d] = (bf16_t)val;
    cv[(kv * S_ + s) * D_ + d] = val;
  }
}

// ---------------------------------------------------------------------------
// Causal flash attention.  One block = (head, 64 q-rows); 32-key chunks,
// online softmax; both GEMMs on bf16 WMMA; Q/K staged via async copies.
// ---------------------------------------------------------------------------
__global__ __launch_bounds__(256) void attn_kernel(
    const bf16_t* __restrict__ qb, const bf16_t* __restrict__ kb,
    const bf16_t* __restrict__ vb, bf16_t* __restrict__ ab)
{
  __shared__ __align__(16) bf16_t Qs[64][136];  // [qrow][D] natural layout
  __shared__ __align__(16) bf16_t Ks[32][136];  // [key][D]  (B^T for scores)
  __shared__ __align__(16) bf16_t Vst[128][40]; // [dim][key] (B^T for P@V)
  __shared__ __align__(16) bf16_t Pb[64][40];   // [qrow][key] (A for P@V)
  __shared__ float Ps[64][33];
  __shared__ float mrow[64], lrow[64], srow[64];

  const int tid  = threadIdx.x;
  const int qblk = blockIdx.x;
  const int h    = blockIdx.y;
  const int kvh  = h >> 2;            // GQA group of 4
  const int q0   = qblk * 64;

  // stage Q once: 64x128 = 1024 chunks of 8 bf16, 4 per thread (async)
  #pragma unroll
  for (int i = 0; i < 4; ++i) {
    int idx = i * 256 + tid;
    int r = idx >> 4, c = idx & 15;          // c: 8-elem chunk in row
    int g = c >> 2, wc = c & 3;              // 32-group, chunk within group
    cp16(&Qs[r][g * 32 + wc * 8],
         &qb[(h * S_ + q0 + r) * D_ + c * 8]);
  }
  if (tid < 64) { mrow[tid] = -1e30f; lrow[tid] = 0.f; }
  async_wait0();

  const int wave = tid >> 5, lane = tid & 31;
  const int n16  = lane & 15, kh = lane >> 4;
  const int pr   = (wave & 3) * 16;
  const int pc   = (wave >> 2) * 16;
  const int oc   = (wave >> 2) * 64;

  v8f z = {0.f,0.f,0.f,0.f,0.f,0.f,0.f,0.f};
  v8f oacc[4];
  for (int t = 0; t < 4; ++t) oacc[t] = z;

  const int jmax = 2 * qblk + 1;
  for (int j = 0; j <= jmax; ++j) {
    const int kbase = j * 32;
    __syncthreads();
    // K chunk 32x128: 512 chunks, 2 per thread (async, natural layout)
    #pragma unroll
    for (int i = 0; i < 2; ++i) {
      int idx = i * 256 + tid;
      int key = idx >> 4, c = idx & 15;
      cp16(&Ks[key][c * 8], &kb[(kvh * S_ + kbase + key) * D_ + c * 8]);
    }
    // V chunk transposed [dim][key] (element-wise; feeds B^T fragments)
    #pragma unroll
    for (int i = 0; i < 16; ++i) {
      int idx = i * 256 + tid;
      int key = idx >> 7, d = idx & 127;
      Vst[d][key] = vb[(kvh * S_ + kbase + key) * D_ + d];
    }
    async_wait0();
    __syncthreads();

    // ---- scores: P(64x32) = Q @ K^T, 4 WMMA k-steps ----
    v8f p = z;
    #pragma unroll
    for (int ks = 0; ks < 4; ++ks) {
      v16bf aq = *(const v16bf*)&Qs[pr + n16][ks * 32 + kh * 16];
      v16bf bk = *(const v16bf*)&Ks[pc + n16][ks * 32 + kh * 16];
      p = wmma_bf16(aq, bk, p);
    }
    #pragma unroll
    for (int v = 0; v < 8; ++v) {
      int row = pr + v + 8 * kh;
      int key = kbase + pc + n16;
      float sc = p[v] * 0.08838834764831845f;   // 1/sqrt(128)
      if (key > q0 + row) sc = -1.0e9f;         // causal mask
      Ps[row][pc + n16] = sc;
    }
    __syncthreads();

    // ---- online softmax per q-row ----
    if (tid < 64) {
      int r = tid;
      float mo = mrow[r], mc = mo;
      for (int c = 0; c < 32; ++c) mc = fmaxf(mc, Ps[r][c]);
      float scale = __expf(mo - mc);
      float ls = 0.f;
      for (int c = 0; c < 32; ++c) {
        float e = __expf(Ps[r][c] - mc);
        ls += e;
        Pb[r][c] = (bf16_t)e;
      }
      mrow[r] = mc;
      lrow[r] = lrow[r] * scale + ls;
      srow[r] = scale;
    }
    __syncthreads();

    // ---- O rescale + accumulate P@V (1 WMMA k-step) ----
    v16bf ap = *(const v16bf*)&Pb[pr + n16][kh * 16];
    #pragma unroll
    for (int t = 0; t < 4; ++t) {
      #pragma unroll
      for (int v = 0; v < 8; ++v)
        oacc[t][v] *= srow[pr + v + 8 * kh];
      v16bf bv = *(const v16bf*)&Vst[oc + t * 16 + n16][kh * 16];
      oacc[t] = wmma_bf16(ap, bv, oacc[t]);
    }
  }
  __syncthreads();

  #pragma unroll
  for (int t = 0; t < 4; ++t)
    #pragma unroll
    for (int v = 0; v < 8; ++v) {
      int row = pr + v + 8 * kh;
      int col = oc + t * 16 + n16;
      float o = oacc[t][v] / lrow[row];
      ab[(q0 + row) * (H_ * D_) + h * D_ + col] = (bf16_t)o;
    }
}

// ---------------------------------------------------------------------------
// Host-side launch.
// Inputs: 0:x 1:cos 2:sin 3:mask 4:wq 5:wk 6:wv 7:wo 8:cache_k 9:cache_v 10:start_pos
// Output: [out (2048*4096) | new_ck (8388608) | new_cv (8388608)] f32.
// ---------------------------------------------------------------------------
extern "C" void kernel_launch(void* const* d_in, const int* in_sizes, int n_in,
                              void* d_out, int out_size, void* d_ws, size_t ws_size,
                              hipStream_t stream)
{
  const float* x     = (const float*)d_in[0];
  const float* cosT  = (const float*)d_in[1];
  const float* sinT  = (const float*)d_in[2];
  const float* wq    = (const float*)d_in[4];
  const float* wk    = (const float*)d_in[5];
  const float* wv    = (const float*)d_in[6];
  const float* wo    = (const float*)d_in[7];
  const float* ck_in = (const float*)d_in[8];
  const float* cv_in = (const float*)d_in[9];

  float* out = (float*)d_out;
  const size_t outN = (size_t)S_ * HID_;
  const size_t ckN  = (size_t)4 * KV_ * 16 * 2048 * 8;
  const size_t cvN  = (size_t)4 * KV_ * 2048 * D_;
  float* ck = out + outN;
  float* cv = ck + ckN;

  char* ws = (char*)d_ws;
  float*  qkv  = (float*)ws;  ws += (size_t)S_ * NQKV * sizeof(float);
  bf16_t* xb   = (bf16_t*)ws; ws += (size_t)S_ * HID_ * sizeof(bf16_t);
  bf16_t* wt   = (bf16_t*)ws; ws += (size_t)NQKV * HID_ * sizeof(bf16_t);
  bf16_t* wot  = (bf16_t*)ws; ws += (size_t)HID_ * HID_ * sizeof(bf16_t);
  bf16_t* qb   = (bf16_t*)ws; ws += (size_t)H_  * S_ * D_ * sizeof(bf16_t);
  bf16_t* kb   = (bf16_t*)ws; ws += (size_t)KV_ * S_ * D_ * sizeof(bf16_t);
  bf16_t* vb   = (bf16_t*)ws; ws += (size_t)KV_ * S_ * D_ * sizeof(bf16_t);
  bf16_t* ab   = (bf16_t*)ws; ws += (size_t)S_  * H_ * D_ * sizeof(bf16_t);

  // Fill full cache outputs from the (zero) input caches; batch-0 slices are
  // overwritten by rope_scatter_kernel.
  (void)hipMemcpyAsync(ck, ck_in, ckN * sizeof(float),
                       hipMemcpyDeviceToDevice, stream);
  (void)hipMemcpyAsync(cv, cv_in, cvN * sizeof(float),
                       hipMemcpyDeviceToDevice, stream);

  // Pre-pass: bf16 conversions + weight transposes (one-time, BW-bound).
  convert_x_kernel<<<(S_ * HID_) / 256, 256, 0, stream>>>(x, xb);
  w_transpose_kernel<<<dim3(4096 / 32, HID_ / 32), 256, 0, stream>>>(
      wq, wt, 4096, HID_);
  w_transpose_kernel<<<dim3(1024 / 32, HID_ / 32), 256, 0, stream>>>(
      wk, wt + (size_t)4096 * HID_, 1024, HID_);
  w_transpose_kernel<<<dim3(1024 / 32, HID_ / 32), 256, 0, stream>>>(
      wv, wt + (size_t)5120 * HID_, 1024, HID_);
  w_transpose_kernel<<<dim3(4096 / 32, HID_ / 32), 256, 0, stream>>>(
      wo, wot, 4096, HID_);

  // QKV projection: [2048 x 6144] = xb @ wt^T
  gemm_bf16_kernel<<<dim3(NQKV / 128, S_ / 128), 256, 0, stream>>>(
      xb, wt, qkv, NQKV);

  int total = S_ * H_ * 64 + S_ * KV_ * 64 + S_ * KV_ * D_;
  rope_scatter_kernel<<<(total + 255) / 256, 256, 0, stream>>>(
      qkv, cosT, sinT, qb, kb, vb, ck, cv);

  attn_kernel<<<dim3(S_ / 64, H_), 256, 0, stream>>>(qb, kb, vb, ab);

  // Output projection: [2048 x 4096] = ab @ wot^T
  gemm_bf16_kernel<<<dim3(HID_ / 128, S_ / 128), 256, 0, stream>>>(
      ab, wot, out, HID_);
}
